// SharedGateRNN_46136538694214
// MI455X (gfx1250) — matrix operand order repaired and checked by reading, not compiled, not run on hardware
//
#include <hip/hip_runtime.h>
#include <hip/hip_bf16.h>
#include <math.h>

// Problem constants (from reference): B=64, T=512, D=256, H=1024
#define RB 64
#define RT 512
#define RD 256
#define RH 1024

typedef __attribute__((ext_vector_type(16))) _Float16 v16h;
typedef __attribute__((ext_vector_type(8)))  _Float16 v8h;
typedef __attribute__((ext_vector_type(8)))  float    v8f;

// ---- WMMA fragment loaders (contiguous-K order, same for A and B; not the ISA table order) ----
// A (16x32 f16): lane l -> row m=l%16; hi=l/16; 16 contiguous K from kc+hi*16
__device__ __forceinline__ v16h load_a_frag(const _Float16* base, int row, int rowstride,
                                            int kc, int hi) {
  return *(const v16h*)(base + (size_t)row * rowstride + kc + hi * 16);
}
// B (32x16 f16): lane l -> col n=l%16; hi=l/16; 16 contiguous K from W[n][kc+hi*16] (B = W^T)
__device__ __forceinline__ v16h load_b_frag(const _Float16* base, int col, int rowstride,
                                            int kc, int hi) {
  return *(const v16h*)(base + (size_t)col * rowstride + kc + hi * 16);
}

// ---- Phase 0: convert weights/x to f16, extract diag(Wh), zero h0 ----
__global__ void k_convert(const float* __restrict__ x, const float* __restrict__ Wx,
                          const float* __restrict__ Wh,
                          _Float16* __restrict__ x16, _Float16* __restrict__ Wx16,
                          _Float16* __restrict__ Wh16, float* __restrict__ whd,
                          float* __restrict__ h0, _Float16* __restrict__ h16_0) {
  int i = blockIdx.x * 256 + threadIdx.x;                 // 0 .. 8388607
  x16[i] = (_Float16)x[i];
  if (i < RH * RH) Wh16[i] = (_Float16)Wh[i];
  if (i < RH * RD) Wx16[i] = (_Float16)Wx[i];
  if (i < RH)      whd[i]  = Wh[(size_t)i * RH + i];
  if (i < RB * RH) { h0[i] = 0.f; h16_0[i] = (_Float16)0.f; }
}

// ---- Phase 0b: a_sx[b*T+t] = x[b,t,:] . Ws + Ws_b (kept fp32-exact, tiny) ----
__global__ void k_asx(const float* __restrict__ x, const float* __restrict__ Ws,
                      const float* __restrict__ Ws_b, float* __restrict__ a_sx) {
  int i = blockIdx.x * 256 + threadIdx.x;                 // 0 .. 32767 (== b*T+t)
  const float* row = x + (size_t)i * RD;
  float acc = Ws_b[0];
#pragma unroll 8
  for (int d = 0; d < RD; ++d) acc += row[d] * Ws[d];
  a_sx[i] = acc;
}

// ---- Phase 1: pre_x = x @ Wx^T + Wx_b, staged into the `gates` output region ----
// Grid: 4096 blocks x 256 thr = 32768 waves; wave = (m-tile, 4 n-tiles)
__global__ void k_gemm_x(const _Float16* __restrict__ x16, const _Float16* __restrict__ Wx16,
                         const float* __restrict__ Wx_b, float* __restrict__ preg) {
  int tid = threadIdx.x, lane = tid & 31, w = tid >> 5;
  int unit = blockIdx.x * 8 + w;                          // 0 .. 32767
  int mt = unit >> 4, ng = unit & 15;
  int m0 = mt * 16, n0 = ng * 64;
  int hi = lane >> 4, lm = lane & 15;
  v8f acc0 = {}, acc1 = {}, acc2 = {}, acc3 = {};
#pragma unroll
  for (int kc = 0; kc < RD; kc += 32) {
    v16h a  = load_a_frag(x16, m0 + lm, RD, kc, hi);
    v16h b0 = load_b_frag(Wx16, n0 +  0 + lm, RD, kc, hi);
    v16h b1 = load_b_frag(Wx16, n0 + 16 + lm, RD, kc, hi);
    v16h b2 = load_b_frag(Wx16, n0 + 32 + lm, RD, kc, hi);
    v16h b3 = load_b_frag(Wx16, n0 + 48 + lm, RD, kc, hi);
    acc0 = __builtin_amdgcn_wmma_f32_16x16x32_f16(false, a, false, b0, (short)0, acc0, false, false);
    acc1 = __builtin_amdgcn_wmma_f32_16x16x32_f16(false, a, false, b1, (short)0, acc1, false, false);
    acc2 = __builtin_amdgcn_wmma_f32_16x16x32_f16(false, a, false, b2, (short)0, acc2, false, false);
    acc3 = __builtin_amdgcn_wmma_f32_16x16x32_f16(false, a, false, b3, (short)0, acc3, false, false);
  }
  v8f* accs[4] = {&acc0, &acc1, &acc2, &acc3};
#pragma unroll
  for (int nt = 0; nt < 4; ++nt) {
    int n = n0 + nt * 16 + lm;
    float bias = Wx_b[n];
#pragma unroll
    for (int r = 0; r < 8; ++r) {
      int m = m0 + ((lane < 16) ? r : r + 8);             // m == b*T + t
      preg[(size_t)m * RH + n] = (*accs[nt])[r] + bias;
    }
  }
}

// ---- Phase 2 (x512): one fused recurrent step ----
// Grid: 32 blocks x 256 thr = 256 waves = 256 (16x16) tiles of [64 x 1024]
__global__ void k_step(int t,
                       const float* __restrict__ hprev, const _Float16* __restrict__ h16prev,
                       float* __restrict__ hnext, _Float16* __restrict__ h16next,
                       const _Float16* __restrict__ Wh16, const float* __restrict__ a_sx,
                       const float* __restrict__ Us, const float* __restrict__ outw,
                       const float* __restrict__ outb, const float* __restrict__ whd,
                       float* __restrict__ ys, float* __restrict__ hs,
                       float* __restrict__ gates, float* __restrict__ leaks,
                       float* __restrict__ rdiags) {
  __shared__ float part_s[4][RB];
  __shared__ float part_y[4][RB];
  __shared__ float s_lds[RB];

  int tid = threadIdx.x;
  // --- scalar gate s[b] = sigmoid(a_sx[b,t] + h_prev[b,:].Us) ; block0 also y_{t-1} ---
  {
    int b = tid & 63, q = tid >> 6;
    const float* hrow = hprev + (size_t)b * RH + q * 256;
    const float* usq  = Us + q * 256;
    float ps = 0.f;
#pragma unroll 8
    for (int k = 0; k < 256; ++k) ps += hrow[k] * usq[k];
    part_s[q][b] = ps;
    if (blockIdx.x == 0) {
      const float* owq = outw + q * 256;
      float py = 0.f;
      if (t > 0) {
#pragma unroll 8
        for (int k = 0; k < 256; ++k) py += hrow[k] * owq[k];
      }
      part_y[q][b] = py;
    }
  }
  __syncthreads();
  if (tid < RB) {
    float a = a_sx[(size_t)tid * RT + t] +
              part_s[0][tid] + part_s[1][tid] + part_s[2][tid] + part_s[3][tid];
    s_lds[tid] = 1.f / (1.f + expf(-a));
    if (blockIdx.x == 0 && t > 0) {
      ys[(size_t)tid * RT + (t - 1)] =
          part_y[0][tid] + part_y[1][tid] + part_y[2][tid] + part_y[3][tid] + outb[0];
    }
  }
  __syncthreads();

  // --- WMMA: C[64,1024] = h_prev @ Wh^T (f16 in, f32 acc), one tile per wave.
  //     Split-K dual accumulators -> two independent WMMA chains per wave so the
  //     XDL pipe isn't stalled by the serial C-accumulate dependency (§7.5). ---
  int lane = tid & 31, w = tid >> 5;
  int unit = blockIdx.x * 8 + w;                          // 0 .. 255
  int mt = unit >> 6, ntile = unit & 63;
  int hi = lane >> 4, lm = lane & 15;
  int m_row = mt * 16 + lm;
  int n = ntile * 16 + lm;
  v8f accA = {}, accB = {};
#pragma unroll 2
  for (int kc = 0; kc < RH; kc += 64) {
    v16h a0 = load_a_frag(h16prev, m_row, RH, kc, hi);
    v16h b0 = load_b_frag(Wh16, n, RH, kc, hi);
    v16h a1 = load_a_frag(h16prev, m_row, RH, kc + 32, hi);
    v16h b1 = load_b_frag(Wh16, n, RH, kc + 32, hi);
    accA = __builtin_amdgcn_wmma_f32_16x16x32_f16(false, a0, false, b0, (short)0, accA, false, false);
    accB = __builtin_amdgcn_wmma_f32_16x16x32_f16(false, a1, false, b1, (short)0, accB, false, false);
  }
  v8f acc = accA + accB;

  // --- fused epilogue: gate, tanh, all five outputs, next-h ---
  float usn = Us[n], whdn = whd[n];
#pragma unroll
  for (int r = 0; r < 8; ++r) {
    int m = mt * 16 + ((lane < 16) ? r : r + 8);          // m == b
    float s = s_lds[m];
    float leak = 1.f - s;
    float sp = s * leak;                                  // sigmoid'
    float hpv = hprev[(size_t)m * RH + n];
    size_t oi = ((size_t)m * RT + t) * RH + n;            // [b,t,n]
    float pre = acc[r] + gates[oi];                       // gates region holds pre_x here
    float ht  = tanhf(pre);
    float h   = leak * hpv + s * ht;
    hs[oi]     = h;
    gates[oi]  = s;                                       // overwrite pre_x after the read
    leaks[oi]  = leak;
    rdiags[oi] = (ht - hpv) * (sp * usn) + s * (1.f - ht * ht) * whdn;
    hnext[(size_t)m * RH + n]   = h;
    h16next[(size_t)m * RH + n] = (_Float16)h;
  }
}

// ---- Phase 3: last y (t = T-1) from final h ----
__global__ void k_final_y(const float* __restrict__ hfin, const float* __restrict__ outw,
                          const float* __restrict__ outb, float* __restrict__ ys) {
  __shared__ float part_y[4][RB];
  int tid = threadIdx.x;
  int b = tid & 63, q = tid >> 6;
  const float* hrow = hfin + (size_t)b * RH + q * 256;
  const float* owq  = outw + q * 256;
  float py = 0.f;
#pragma unroll 8
  for (int k = 0; k < 256; ++k) py += hrow[k] * owq[k];
  part_y[q][b] = py;
  __syncthreads();
  if (tid < RB)
    ys[(size_t)tid * RT + (RT - 1)] =
        part_y[0][tid] + part_y[1][tid] + part_y[2][tid] + part_y[3][tid] + outb[0];
}

extern "C" void kernel_launch(void* const* d_in, const int* in_sizes, int n_in,
                              void* d_out, int out_size, void* d_ws, size_t ws_size,
                              hipStream_t stream) {
  (void)in_sizes; (void)n_in; (void)out_size; (void)ws_size;

  const float* x    = (const float*)d_in[0];   // [B,T,D]
  const float* Wx_w = (const float*)d_in[1];   // [H,D]
  const float* Wx_b = (const float*)d_in[2];   // [H]
  const float* Wh_w = (const float*)d_in[3];   // [H,H]
  const float* Ws_w = (const float*)d_in[4];   // [1,D]
  const float* Ws_b = (const float*)d_in[5];   // [1]
  const float* Us_w = (const float*)d_in[6];   // [1,H]
  const float* ow   = (const float*)d_in[7];   // [1,H]
  const float* ob   = (const float*)d_in[8];   // [1]

  // Output layout: ys [B,T,1] | hs,gates,leaks,rdiags [B,T,H]
  float* out    = (float*)d_out;
  float* ys     = out;
  float* hs     = out + (size_t)RB * RT;
  float* gates  = hs     + (size_t)RB * RT * RH;
  float* leaks  = gates  + (size_t)RB * RT * RH;
  float* rdiags = leaks  + (size_t)RB * RT * RH;

  // Workspace carve (~20.3 MB, all offsets >=1KB aligned)
  char* ws = (char*)d_ws;
  float*    a_sx  = (float*)(ws + 0);                         // 32768 f32
  float*    whd   = (float*)(ws + 131072);                    // 1024 f32
  float*    hbuf0 = (float*)(ws + 135168);                    // 64*1024 f32
  float*    hbuf1 = (float*)(ws + 397312);                    // 64*1024 f32
  _Float16* h16b0 = (_Float16*)(ws + 659456);                 // 64*1024 f16
  _Float16* h16b1 = (_Float16*)(ws + 790528);                 // 64*1024 f16
  _Float16* Wx16  = (_Float16*)(ws + 921600);                 // H*D f16
  _Float16* Wh16  = (_Float16*)(ws + 1445888);                // H*H f16
  _Float16* x16   = (_Float16*)(ws + 3543040);                // B*T*D f16

  float*    hb[2]  = {hbuf0, hbuf1};
  _Float16* h16[2] = {h16b0, h16b1};

  // Phase 0: conversions + zero-init h0
  k_convert<<<(RB * RT * RD) / 256, 256, 0, stream>>>(x, Wx_w, Wh_w, x16, Wx16, Wh16,
                                                      whd, hbuf0, h16b0);
  // Phase 0b: scalar-gate input projection (exact fp32)
  k_asx<<<(RB * RT) / 256, 256, 0, stream>>>(x, Ws_w, Ws_b, a_sx);
  // Phase 1: big WMMA GEMM pre_x -> staged into `gates`
  k_gemm_x<<<4096, 256, 0, stream>>>(x16, Wx16, Wx_b, gates);
  // Phase 2: 512 fused recurrent steps (graph-replayed small nodes)
  for (int t = 0; t < RT; ++t) {
    k_step<<<32, 256, 0, stream>>>(t, hb[t & 1], h16[t & 1], hb[(t + 1) & 1],
                                   h16[(t + 1) & 1], Wh16, a_sx, Us_w, ow, ob, whd,
                                   ys, hs, gates, leaks, rdiags);
  }
  // Phase 3: last y from final h (lands back in buffer 0 after 512 steps)
  k_final_y<<<1, 256, 0, stream>>>(hb[0], ow, ob, ys);
}
